// NeuralVolatitlityModel_85160611545171
// MI455X (gfx1250) — compile-verified, matrix operand order checked by reading, not run on hardware
//
#include <hip/hip_runtime.h>
#include <hip/hip_bf16.h>
#include <math.h>

typedef __attribute__((ext_vector_type(16))) _Float16 v16h;
typedef __attribute__((ext_vector_type(8)))  float    v8f;

// Problem constants
#define NSTEP 256
#define ROWS  256
#define DH    256
#define DIN   64
#define DZ    64

// ---- d_ws byte layout (all 256B-aligned) ----
constexpr size_t OFF_W1  = 0;        // W_xh_ih  256x64   f16  (32768 B)
constexpr size_t OFF_W1H = 32768;    // W_xh_hh  256x256  f16  (131072 B)
constexpr size_t OFF_W2  = 163840;   // W_hz_ih  256x320  f16  (163840 B)
constexpr size_t OFF_W2H = 327680;   // W_hz_hh  256x256  f16
constexpr size_t OFF_W3  = 458752;   // W_zh_ih  256x128  f16  (65536 B)
constexpr size_t OFF_W3H = 524288;   // W_zh_hh  256x256  f16
constexpr size_t OFF_WZM = 655360;   // W_zm     64x256   f16  (32768 B)
constexpr size_t OFF_WZL = 688128;   // W_zl     64x256   f16
constexpr size_t OFF_WXM = 720896;   // W_xm     64x256   f16
constexpr size_t OFF_HO  = 753664;   // ho state 256x256  f16
constexpr size_t OFF_XT  = 884736;   // xt       256x64   f16
constexpr size_t OFF_PX  = 917504;   // prev_x   256x64   f16
constexpr size_t OFF_B1  = 950272;   // b_xh_ih + b_xh_hh  256 f32
constexpr size_t OFF_B2  = 951296;   // b_hz_ih + b_hz_hh  256 f32
constexpr size_t OFF_B3  = 952320;   // b_zh_ih + b_zh_hh  256 f32

constexpr unsigned SMEM_BYTES = (65536 + 65536 + 16384) * 2; // hi, hl, z in f16 = 294912 B

// ---------- prep kernels ----------
__global__ void cvt_f32_f16(const float* __restrict__ s, _Float16* __restrict__ d, int n) {
    int i = blockIdx.x * blockDim.x + threadIdx.x;
    if (i < n) d[i] = (_Float16)s[i];
}
__global__ void bias_sum(const float* __restrict__ a, const float* __restrict__ b,
                         float* __restrict__ d, int n) {
    int i = blockIdx.x * blockDim.x + threadIdx.x;
    if (i < n) d[i] = a[i] + b[i];
}

// ---------- WMMA helpers ----------
// C += A * W^T for NT adjacent 16x16 tiles on one wave.
// A: row-major [*, lda] f16, rows r0..r0+15 used.  W: row-major [*, ldw] f16,
// rows c0..c0+16*NT-1 used (these are the OUTPUT columns).  K multiple of 32.
// f16 fragments as loaded here (same k order for A and B): lane L -> row (L&15), contiguous 16 halfs at k + ((L>>4)<<4).
template <int NT>
__device__ __forceinline__ void wmma_acc(v8f* acc,
    const _Float16* __restrict__ A, int lda,
    const _Float16* __restrict__ W, int ldw,
    int r0, int c0, int K, int lane)
{
    const int r  = lane & 15;
    const int kq = (lane >> 4) << 4;
    for (int k = 0; k < K; k += 32) {
        v16h a = *(const v16h*)(A + (size_t)(r0 + r) * lda + k + kq);
#pragma unroll
        for (int nt = 0; nt < NT; ++nt) {
            v16h b = *(const v16h*)(W + (size_t)(c0 + nt * 16 + r) * ldw + k + kq);
            acc[nt] = __builtin_amdgcn_wmma_f32_16x16x32_f16(
                false, a, false, b, (short)0, acc[nt], false, false);
        }
    }
}

// D-layout: lane L, vgpr v -> element (m = v + ((L>>4)<<3), n = L&15) of the tile.
template <int NT>
__device__ __forceinline__ void commit_tanh(const v8f* acc, _Float16* H, int ldh,
                                            const float* __restrict__ bias,
                                            int r0, int c0, int lane)
{
    const int nl = lane & 15;
    const int mh = (lane >> 4) << 3;
#pragma unroll
    for (int td = 0; td < NT; ++td) {
        const int n = c0 + td * 16 + nl;
        const float b = bias[n];
#pragma unroll
        for (int v = 0; v < 8; ++v) {
            const int m = r0 + mh + v;
            H[(size_t)m * ldh + n] = (_Float16)tanhf(acc[td][v] + b);
        }
    }
}

// ---------- main persistent recurrent kernel: 1 block, 32 waves ----------
__global__ __launch_bounds__(1024) void vrnn_main(
    const float* __restrict__ x,        // [256,256,64]
    const float* __restrict__ h_in0,    // [256,256]
    const float* __restrict__ h_lat0,   // [256,256]
    const float* __restrict__ z0,       // [256,64]
    const float* __restrict__ tmp0,     // [256,64]
    const float* __restrict__ noise_z,  // [256,256,64]
    const float* __restrict__ noise_x,  // [256,256,64]
    const float* __restrict__ b_zm,     // [64]
    const float* __restrict__ b_zl,     // [64]
    const float* __restrict__ b_xm,     // [64]
    char* __restrict__ ws,
    float* __restrict__ out)            // [256,256,64]
{
    extern __shared__ _Float16 smem[];
    _Float16* HI = smem;            // 256x256
    _Float16* HL = smem + 65536;    // 256x256
    _Float16* ZS = smem + 131072;   // 256x64

    const _Float16* W1  = (const _Float16*)(ws + OFF_W1);
    const _Float16* W1H = (const _Float16*)(ws + OFF_W1H);
    const _Float16* W2  = (const _Float16*)(ws + OFF_W2);
    const _Float16* W2H = (const _Float16*)(ws + OFF_W2H);
    const _Float16* W3  = (const _Float16*)(ws + OFF_W3);
    const _Float16* W3H = (const _Float16*)(ws + OFF_W3H);
    const _Float16* WZM = (const _Float16*)(ws + OFF_WZM);
    const _Float16* WZL = (const _Float16*)(ws + OFF_WZL);
    const _Float16* WXM = (const _Float16*)(ws + OFF_WXM);
    _Float16* HO = (_Float16*)(ws + OFF_HO);
    _Float16* XT = (_Float16*)(ws + OFF_XT);
    _Float16* PX = (_Float16*)(ws + OFF_PX);
    const float* B1 = (const float*)(ws + OFF_B1);
    const float* B2 = (const float*)(ws + OFF_B2);
    const float* B3 = (const float*)(ws + OFF_B3);

    const int tid  = threadIdx.x;
    const int lane = tid & 31;
    const int w    = tid >> 5;
    const int r0   = (w >> 1) << 4;   // 16 row tiles, 2 waves each
    const int ch   = w & 1;           // column half (128 cols) / pair (32 cols)

    // Initialize recurrent LDS state from fp32 inputs.
    for (int i = tid; i < 65536; i += 1024) {
        HI[i] = (_Float16)h_in0[i];
        HL[i] = (_Float16)h_lat0[i];
    }
    for (int i = tid; i < 16384; i += 1024) ZS[i] = (_Float16)z0[i];
    __syncthreads();

    for (int t = 0; t < NSTEP; ++t) {
        // ---- phase A: stage xt = x[t], px = (t ? x[t-1] : tmp0) as f16 ----
        {
            const float* xsrc = x + (size_t)t * (ROWS * DIN);
            const float* psrc = (t == 0) ? tmp0 : (x + (size_t)(t - 1) * (ROWS * DIN));
            for (int i = tid; i < ROWS * DIN; i += 1024) {
                XT[i] = (_Float16)xsrc[i];
                PX[i] = (_Float16)psrc[i];
            }
            __syncthreads();
        }

        // ---- stage 1: hi = tanh(xt*W1^T + hi*W1h^T + B1) ----
        {
            v8f acc[8] = {};
            wmma_acc<8>(acc, XT, DIN, W1,  DIN, r0, ch * 128, DIN, lane);
            wmma_acc<8>(acc, HI, DH,  W1H, DH,  r0, ch * 128, DH,  lane);
            __syncthreads();                       // all reads of old HI done
            commit_tanh<8>(acc, HI, DH, B1, r0, ch * 128, lane);
            __syncthreads();
        }

        // ---- stage 2: hl = tanh([hi,z]*W2^T + hl*W2h^T + B2) ----
        {
            v8f acc[8] = {};
            wmma_acc<8>(acc, HI, DH, W2,       320, r0, ch * 128, DH, lane);
            wmma_acc<8>(acc, ZS, DZ, W2 + 256, 320, r0, ch * 128, DZ, lane);
            wmma_acc<8>(acc, HL, DH, W2H,      DH,  r0, ch * 128, DH, lane);
            __syncthreads();
            commit_tanh<8>(acc, HL, DH, B2, r0, ch * 128, lane);
            __syncthreads();
        }

        // ---- stage 3: z = exp(hl*Wzl^T + b_zl) * noise_z + (hl*Wzm^T + b_zm) ----
        {
            v8f am[2] = {};
            v8f al[2] = {};
            wmma_acc<2>(am, HL, DH, WZM, DH, r0, ch * 32, DH, lane);
            wmma_acc<2>(al, HL, DH, WZL, DH, r0, ch * 32, DH, lane);
            __syncthreads();
            const int nl = lane & 15, mh = (lane >> 4) << 3;
#pragma unroll
            for (int td = 0; td < 2; ++td) {
                const int n = ch * 32 + td * 16 + nl;
                const float bm = b_zm[n], bl = b_zl[n];
#pragma unroll
                for (int v = 0; v < 8; ++v) {
                    const int m = r0 + mh + v;
                    const float mean = am[td][v] + bm;
                    const float lv   = al[td][v] + bl;
                    const float nz   = noise_z[((size_t)t * ROWS + m) * DZ + n];
                    ZS[m * DZ + n] = (_Float16)(expf(lv) * nz + mean);
                }
            }
            __syncthreads();
        }

        // ---- stage 4: ho = tanh([z,px]*W3^T + ho*W3h^T + B3) ----
        {
            v8f acc[8] = {};
            wmma_acc<8>(acc, ZS, DZ,  W3,      128, r0, ch * 128, DZ,  lane);
            wmma_acc<8>(acc, PX, DIN, W3 + 64, 128, r0, ch * 128, DIN, lane);
            wmma_acc<8>(acc, HO, DH,  W3H,     DH,  r0, ch * 128, DH,  lane);
            __syncthreads();
            commit_tanh<8>(acc, HO, DH, B3, r0, ch * 128, lane);
            __syncthreads();
        }

        // ---- stage 5: mean_x = ho*Wxm^T + b_xm; out = exp(mean_x)*noise_x + mean_x ----
        {
            v8f ax[2] = {};
            wmma_acc<2>(ax, HO, DH, WXM, DH, r0, ch * 32, DH, lane);
            const int nl = lane & 15, mh = (lane >> 4) << 3;
#pragma unroll
            for (int td = 0; td < 2; ++td) {
                const int n = ch * 32 + td * 16 + nl;
                const float b = b_xm[n];
#pragma unroll
                for (int v = 0; v < 8; ++v) {
                    const int m = r0 + mh + v;
                    const float mx = ax[td][v] + b;
                    const float nx = noise_x[((size_t)t * ROWS + m) * DIN + n];
                    out[((size_t)t * ROWS + m) * DIN + n] = expf(mx) * nx + mx;
                }
            }
            // no barrier needed: next phase A writes XT/PX, which stage 5 never reads
        }
    }
}

// ---------- host entry ----------
extern "C" void kernel_launch(void* const* d_in, const int* in_sizes, int n_in,
                              void* d_out, int out_size, void* d_ws, size_t ws_size,
                              hipStream_t stream) {
    (void)in_sizes; (void)n_in; (void)out_size; (void)ws_size;
    char* ws = (char*)d_ws;

    auto cvt = [&](int idx, size_t off, int n) {
        cvt_f32_f16<<<(n + 255) / 256, 256, 0, stream>>>(
            (const float*)d_in[idx], (_Float16*)(ws + off), n);
    };
    // weights -> f16
    cvt(8,  OFF_W1,  ROWS * DIN);        // W_xh_ih
    cvt(10, OFF_W1H, ROWS * DH);         // W_xh_hh
    cvt(12, OFF_W2,  ROWS * (DH + DZ));  // W_hz_ih
    cvt(14, OFF_W2H, ROWS * DH);         // W_hz_hh
    cvt(16, OFF_W3,  ROWS * (DZ + DIN)); // W_zh_ih
    cvt(18, OFF_W3H, ROWS * DH);         // W_zh_hh
    cvt(20, OFF_WZM, DZ * DH);           // W_zm
    cvt(22, OFF_WZL, DZ * DH);           // W_zl
    cvt(24, OFF_WXM, DIN * DH);          // W_xm
    // initial ho state -> f16
    cvt(3,  OFF_HO,  ROWS * DH);         // h_out0
    // fused biases (b_ih + b_hh)
    bias_sum<<<1, 256, 0, stream>>>((const float*)d_in[9],  (const float*)d_in[11],
                                    (float*)(ws + OFF_B1), DH);
    bias_sum<<<1, 256, 0, stream>>>((const float*)d_in[13], (const float*)d_in[15],
                                    (float*)(ws + OFF_B2), DH);
    bias_sum<<<1, 256, 0, stream>>>((const float*)d_in[17], (const float*)d_in[19],
                                    (float*)(ws + OFF_B3), DH);

    vrnn_main<<<1, 1024, SMEM_BYTES, stream>>>(
        (const float*)d_in[0],  // x
        (const float*)d_in[1],  // h_in0
        (const float*)d_in[2],  // h_lat0
        (const float*)d_in[4],  // z0
        (const float*)d_in[5],  // tmp0
        (const float*)d_in[6],  // noise_z
        (const float*)d_in[7],  // noise_x
        (const float*)d_in[21], // b_zm
        (const float*)d_in[23], // b_zl
        (const float*)d_in[25], // b_xm
        ws,
        (float*)d_out);
}
